// EvoformerIteration_73254962201342
// MI455X (gfx1250) — matrix operand order repaired and checked by reading, not compiled, not run on hardware
//
#include <hip/hip_runtime.h>

// ---------------------------------------------------------------------------
// Types for CDNA5 WMMA (wave32): v16bf = 16 x bf16 (8 VGPRs), v8f = f32 accum
// ---------------------------------------------------------------------------
typedef __attribute__((ext_vector_type(16))) __bf16 v16bf;
typedef __attribute__((ext_vector_type(8)))  float  v8f;

__device__ __forceinline__ unsigned short f2bf(float f) {
  unsigned u = __float_as_uint(f);
  u += 0x7fffu + ((u >> 16) & 1u);          // round-to-nearest-even
  return (unsigned short)(u >> 16);
}
__device__ __forceinline__ float bf2f(unsigned short s) {
  return __uint_as_float(((unsigned)s) << 16);
}

union FragU { v16bf v; unsigned u[8]; };

// A-matrix 16x32 bf16 fragment: lane half selects contiguous K range of 16
// (lanes 0-15: K 0..15, lanes 16-31: K 16..31). p = row base + (laneHalf*16).
__device__ __forceinline__ v16bf load_fragA(const unsigned short* p) {
  FragU f;
  const unsigned* a = (const unsigned*)p;
  const unsigned* b = (const unsigned*)(p + 8);
  f.u[0] = a[0]; f.u[1] = a[1]; f.u[2] = a[2]; f.u[3] = a[3];
  f.u[4] = b[0]; f.u[5] = b[1]; f.u[6] = b[2]; f.u[7] = b[3];
  return f.v;
}
// B-matrix 32x16 bf16 fragment: lane holds column N, contiguous K range of 16
// (lanes 0-15: K 0..15, lanes 16-31: K 16..31). p = N-row base + laneHalf*16.
__device__ __forceinline__ v16bf load_fragB(const unsigned short* p) {
  FragU f;
  const unsigned* a = (const unsigned*)p;
#pragma unroll
  for (int i = 0; i < 8; ++i) f.u[i] = a[i];
  return f.v;
}

// gfx1250 async global->LDS copy, 16B per lane; tracked by ASYNCcnt.
__device__ __forceinline__ void async_ld16(unsigned lds_off, const unsigned short* g) {
  asm volatile("global_load_async_to_lds_b128 %0, %1, off"
               :: "v"(lds_off), "v"((unsigned long long)(__SIZE_TYPE__)g)
               : "memory");
}
__device__ __forceinline__ void wait_async0() {
  asm volatile("s_wait_asynccnt 0x0" ::: "memory");
}

// Shared epilogue for one 16x16 tile held in a v8f accumulator.
//   wmode 0: normal store (optional f32 + bf16 outputs, optional residual)
//   wmode 1: OPM scatter  -> bf16 [(gi/32)*192+(gj/32)][(gi%32)*32 + gj%32]
//   wmode 2: tri einsum   -> f32  [(gi*192+gj)*128 + batch]
__device__ __forceinline__ void ep_store(v8f acc, int ti, int tj, int lane,
                                         float* __restrict__ Cf,
                                         unsigned short* __restrict__ Cb,
                                         const float* __restrict__ bias,
                                         const float* __restrict__ resid,
                                         int N, int b, long long sC,
                                         int act, float scale, int wmode) {
  int ln15 = lane & 15, lh = lane >> 4;
  int col  = tj * 16 + ln15;
  int row0 = ti * 16 + (lh << 3);
  float bv = bias ? bias[col] : 0.f;
#pragma unroll
  for (int r = 0; r < 8; ++r) {
    float v = acc[r] * scale + bv;
    if (act == 1)      v = v > 0.f ? v : 0.f;            // relu
    else if (act == 2) v = 1.f / (1.f + __expf(-v));     // sigmoid
    long long grow = row0 + r;
    if (wmode == 0) {
      long long idx = (long long)b * sC + grow * N + col;
      if (Cb) Cb[idx] = f2bf(v);
      if (Cf) { if (resid) v += resid[idx]; Cf[idx] = v; }
    } else if (wmode == 1) {
      long long idx = (((grow >> 5) * 192 + (col >> 5)) << 10) +
                      ((grow & 31) << 5) + (col & 31);
      Cb[idx] = f2bf(v);
    } else {
      long long idx = (grow * 192 + col) * 128 + b;
      Cf[idx] = v;
    }
  }
}

// ---------------------------------------------------------------------------
// LDS-staged, double-buffered WMMA GEMM:  C[b] = act(scale*A[b]@B[b]^T + bias)
//   A bf16 [nb][M][K] row-major, B bf16 [nb][N][K] (B^T row-major)
//   Requires M%64==0, N%64==0, K%32==0. 256 threads = 8 waves (4x2);
//   macro tile 64x64; per wave: one A-frag x two B-frags = 2 WMMA / k-step.
//   k-slices staged via global_load_async_to_lds_b128 (ASYNCcnt) and
//   consumed from LDS; next slice issued before computing current (overlap).
// ---------------------------------------------------------------------------
#define LDP 40  // padded LDS row stride (elements): 80B rows, 16B aligned segs

__global__ void __launch_bounds__(256)
k_gemm_wmma_tiled(const unsigned short* __restrict__ A,
                  const unsigned short* __restrict__ B,
                  float* __restrict__ Cf, unsigned short* __restrict__ Cb,
                  const float* __restrict__ bias, const float* __restrict__ resid,
                  int M, int N, int K, int nb,
                  long long sA, long long sB, long long sC,
                  int act, float scale, int wmode) {
  __shared__ unsigned short shA[2][64 * LDP];
  __shared__ unsigned short shB[2][64 * LDP];
  int tid = threadIdx.x;
  int bn = N >> 6;
  int per = (M >> 6) * bn;
  long long blk = blockIdx.x;
  int b  = (int)(blk / per);
  int t  = (int)(blk - (long long)b * per);
  int gm = t / bn, gn = t - gm * bn;

  const unsigned short* Ab = A + (long long)b * sA + ((long long)gm << 6) * K;
  const unsigned short* Bb = B + (long long)b * sB + ((long long)gn << 6) * K;

  int r = tid >> 2, s = tid & 3;   // staging: 64 rows x 4 16B segments
  unsigned aoff = (unsigned)(__SIZE_TYPE__)(&shA[0][0]) + (unsigned)(r * LDP + s * 8) * 2;
  unsigned boff = (unsigned)(__SIZE_TYPE__)(&shB[0][0]) + (unsigned)(r * LDP + s * 8) * 2;
  const unsigned BUFSTRIDE = 64 * LDP * 2;  // bytes per buffer
  long long grow = (long long)r * K + s * 8;

  int nkb = K >> 5;
  // stage slice 0 into buffer 0
  async_ld16(aoff, Ab + grow);
  async_ld16(boff, Bb + grow);

  int wv = tid >> 5, lane = tid & 31;
  int wr = wv >> 1, wc = wv & 1;   // 4 x 2 wave grid
  int ln15 = lane & 15, lh = lane >> 4;
  const unsigned short* saBase = &shA[0][(wr * 16 + ln15) * LDP + (lh << 4)];
  const unsigned short* sbBase = &shB[0][(wc * 32 + ln15) * LDP + (lh << 4)];

  v8f acc0 = {}, acc1 = {};
  wait_async0();
  __syncthreads();
  for (int kb = 0; kb < nkb; ++kb) {
    int cur = kb & 1;
    if (kb + 1 < nkb) {             // issue next slice into other buffer
      unsigned bo = (unsigned)((cur ^ 1) * BUFSTRIDE);
      long long go = grow + ((long long)(kb + 1) << 5);
      async_ld16(aoff + bo, Ab + go);
      async_ld16(boff + bo, Bb + go);
    }
    const unsigned short* sa  = saBase + cur * (64 * LDP);
    const unsigned short* sb0 = sbBase + cur * (64 * LDP);
    v16bf af  = load_fragA(sa);
    v16bf bf0 = load_fragB(sb0);
    v16bf bf1 = load_fragB(sb0 + 16 * LDP);
    acc0 = __builtin_amdgcn_wmma_f32_16x16x32_bf16(false, af, false, bf0,
                                                   (short)0, acc0, false, false);
    acc1 = __builtin_amdgcn_wmma_f32_16x16x32_bf16(false, af, false, bf1,
                                                   (short)0, acc1, false, false);
    if (kb + 1 < nkb) { wait_async0(); __syncthreads(); }
  }
  int ti  = (gm << 2) + wr;
  int tj0 = (gn << 2) + (wc << 1);
  ep_store(acc0, ti, tj0,     lane, Cf, Cb, bias, resid, N, b, sC, act, scale, wmode);
  ep_store(acc1, ti, tj0 + 1, lane, Cf, Cb, bias, resid, N, b, sC, act, scale, wmode);
}

// ---------------------------------------------------------------------------
// Direct-load WMMA GEMM (one wave per 16x16 tile) for shapes where the tiled
// kernel's 64x64 macro tile doesn't divide (N=32 attention PV GEMMs).
// ---------------------------------------------------------------------------
__global__ void k_gemm_wmma(const unsigned short* __restrict__ A,
                            const unsigned short* __restrict__ B,
                            float* __restrict__ Cf, unsigned short* __restrict__ Cb,
                            const float* __restrict__ bias,
                            const float* __restrict__ resid,
                            int M, int N, int K, int nb,
                            long long sA, long long sB, long long sC,
                            int act, float scale, int wmode) {
  long long wave = ((long long)blockIdx.x * blockDim.x + threadIdx.x) >> 5;
  int lane = threadIdx.x & 31;
  int tn = N >> 4;
  int tilesPer = (M >> 4) * tn;
  long long total = (long long)tilesPer * nb;
  if (wave >= total) return;                 // whole-wave exit: EXEC stays full
  int b  = (int)(wave / tilesPer);
  int t  = (int)(wave - (long long)b * tilesPer);
  int ti = t / tn, tj = t - ti * tn;
  int ln15 = lane & 15, lh = lane >> 4;

  const unsigned short* Ap = A + (long long)b * sA + (long long)(ti * 16 + ln15) * K + (lh << 4);
  const unsigned short* Bp = B + (long long)b * sB + (long long)(tj * 16 + ln15) * K + (lh << 4);

  v8f acc = {};
  for (int k = 0; k < K; k += 32) {
    if (k + 32 < K) {
      __builtin_prefetch(Ap + k + 32, 0, 1);  // global_prefetch_b8
      __builtin_prefetch(Bp + k + 32, 0, 1);
    }
    v16bf af  = load_fragA(Ap + k);
    v16bf bfm = load_fragB(Bp + k);
    acc = __builtin_amdgcn_wmma_f32_16x16x32_bf16(false, af, false, bfm,
                                                  (short)0, acc, false, false);
  }
  ep_store(acc, ti, tj, lane, Cf, Cb, bias, resid, N, b, sC, act, scale, wmode);
}

// f32 [K][N] weight -> bf16 [N][K] (transposed for fragB loads)
__global__ void k_conv_t(const float* __restrict__ w, unsigned short* __restrict__ wt,
                         int K, int N) {
  long long i = (long long)blockIdx.x * blockDim.x + threadIdx.x;
  if (i >= (long long)K * N) return;
  int k = (int)(i / N), n = (int)(i % N);
  wt[(long long)n * K + k] = f2bf(w[i]);
}

// LayerNorm over last dim; one wave per row; writes bf16 for downstream WMMA.
__global__ void k_ln_rows(const float* __restrict__ x, unsigned short* __restrict__ out,
                          const float* __restrict__ g, const float* __restrict__ b,
                          int rows, int C) {
  int row = blockIdx.x * (blockDim.x >> 5) + (threadIdx.x >> 5);
  if (row >= rows) return;
  int lane = threadIdx.x & 31;
  const float* xr = x + (long long)row * C;
  float s = 0.f, s2 = 0.f;
  for (int c = lane; c < C; c += 32) { float v = xr[c]; s += v; s2 += v * v; }
  for (int o = 16; o; o >>= 1) { s += __shfl_xor(s, o, 32); s2 += __shfl_xor(s2, o, 32); }
  float mean = s / C, var = s2 / C - mean * mean;
  float inv = rsqrtf(var + 1e-5f);
  unsigned short* orow = out + (long long)row * C;
  for (int c = lane; c < C; c += 32)
    orow[c] = f2bf((xr[c] - mean) * inv * g[c] + b[c]);
}

// Split fused qkv (bf16 [R][768], layout (c,3h) = c*24 + {h, 8+h, 16+h}) into
//   Q  bf16 [P*8][rows][32]  (scaled by qscale)
//   K  bf16 [P*8][rows][32]
//   Vt bf16 [P*8][32][rows]  (transposed so PV GEMM B-operand is [N=c][K=row])
// mode 0: global row = p*192 + q ; mode 1: global row = q*192 + p
__global__ void k_split_heads(const unsigned short* __restrict__ qkv,
                              unsigned short* __restrict__ Q,
                              unsigned short* __restrict__ Kk,
                              unsigned short* __restrict__ Vt,
                              int P, int rows, int mode, float qscale) {
  long long i = (long long)blockIdx.x * blockDim.x + threadIdx.x;
  if (i >= (long long)P * rows * 256) return;
  int c = (int)(i & 31);
  int h = (int)((i >> 5) & 7);
  long long pq = i >> 8;
  int q = (int)(pq % rows);
  int p = (int)(pq / rows);
  long long grow = (mode == 0) ? ((long long)p * 192 + q) : ((long long)q * 192 + p);
  const unsigned short* s = qkv + grow * 768 + (long long)c * 24 + h;
  long long bb = (long long)p * 8 + h;
  Q [(bb * rows + q) * 32 + c] = f2bf(bf2f(s[0]) * qscale);
  Kk[(bb * rows + q) * 32 + c] = s[8];
  Vt[(bb * 32 + c) * rows + q] = s[16];
}

// out[r*8+h] = scale * sum_c x[r][c] * w[c][h]   (N=8 too small for WMMA tile)
__global__ void k_dot8_f32(const float* __restrict__ x, const float* __restrict__ w,
                           float* __restrict__ out, int rows, int C, float scale) {
  long long i = (long long)blockIdx.x * blockDim.x + threadIdx.x;
  if (i >= (long long)rows * 8) return;
  int h = (int)(i & 7);
  const float* xr = x + (i >> 3) * C;
  float s = 0.f;
  for (int c = 0; c < C; ++c) s += xr[c] * w[c * 8 + h];
  out[i] = s * scale;
}
__global__ void k_dot8_bf16(const unsigned short* __restrict__ x, const float* __restrict__ w,
                            float* __restrict__ out, int rows, int C, float scale) {
  long long i = (long long)blockIdx.x * blockDim.x + threadIdx.x;
  if (i >= (long long)rows * 8) return;
  int h = (int)(i & 7);
  const unsigned short* xr = x + (i >> 3) * C;
  float s = 0.f;
  for (int c = 0; c < C; ++c) s += bf2f(xr[c]) * w[c * 8 + h];
  out[i] = s * scale;
}

// In-place softmax over cols of bf16 [nb][rows][cols] + optional strided bias:
//   bias index = row*rs + col*cs + (batch & 7).   cols in {64,192}.
__global__ void k_softmax(unsigned short* __restrict__ aff, const float* __restrict__ bias,
                          int nb, int rows, int cols, int rs, int cs) {
  long long wave = ((long long)blockIdx.x * blockDim.x + threadIdx.x) >> 5;
  if (wave >= (long long)nb * rows) return;
  int lane = threadIdx.x & 31;
  int b   = (int)(wave / rows);
  int row = (int)(wave - (long long)b * rows);
  int h = b & 7;
  unsigned short* a = aff + ((long long)b * rows + row) * cols;
  int nv = cols >> 5;
  float vals[6];
  float mx = -3.0e38f;
  for (int t = 0; t < nv; ++t) {
    int c = lane + (t << 5);
    float v = bf2f(a[c]);
    if (bias) v += bias[(long long)row * rs + (long long)c * cs + h];
    vals[t] = v; mx = fmaxf(mx, v);
  }
  for (int o = 16; o; o >>= 1) mx = fmaxf(mx, __shfl_xor(mx, o, 32));
  float sum = 0.f;
  for (int t = 0; t < nv; ++t) { float e = __expf(vals[t] - mx); vals[t] = e; sum += e; }
  for (int o = 16; o; o >>= 1) sum += __shfl_xor(sum, o, 32);
  float inv = 1.f / sum;
  for (int t = 0; t < nv; ++t) a[lane + (t << 5)] = f2bf(vals[t] * inv);
}

// merged[grow][c*8+h] = attnout[(p*8+h)][q][c] * gate[grow][c*8+h]  (bf16 out)
__global__ void k_merge_gate(const float* __restrict__ ao, const float* __restrict__ gate,
                             unsigned short* __restrict__ out, int P, int rows, int mode) {
  long long i = (long long)blockIdx.x * blockDim.x + threadIdx.x;
  if (i >= (long long)P * 8 * rows * 32) return;
  int c = (int)(i & 31);
  long long j = i >> 5;
  int q = (int)(j % rows);
  long long k = j / rows;
  int h = (int)(k & 7);
  int p = (int)(k >> 3);
  long long grow = (mode == 0) ? ((long long)p * 192 + q) : ((long long)q * 192 + p);
  long long oi = grow * 256 + c * 8 + h;
  out[oi] = f2bf(ao[i] * gate[oi]);
}

// tri-mul gate+transpose: in A,S f32 [36864][128] (S pre-sigmoided);
// out bf16 [c][192][192]: mode 0 -> [c][r1][r2], mode 1 -> [c][r2][r1]
__global__ void k_mul_t(const float* __restrict__ A, const float* __restrict__ S,
                        unsigned short* __restrict__ out, int mode) {
  long long i = (long long)blockIdx.x * blockDim.x + threadIdx.x;
  if (i >= 36864LL * 128) return;
  int c = (int)(i & 127);
  long long r = i >> 7;
  int r2 = (int)(r % 192);
  int r1 = (int)(r / 192);
  float v = A[i] * S[i];
  long long oi = (mode == 0) ? ((long long)c * 36864 + (long long)r1 * 192 + r2)
                             : ((long long)c * 36864 + (long long)r2 * 192 + r1);
  out[oi] = f2bf(v);
}

__global__ void k_ew_mac(float* __restrict__ x, const float* __restrict__ a,
                         const float* __restrict__ b, long long n) {
  long long i = (long long)blockIdx.x * blockDim.x + threadIdx.x;
  if (i >= n) return;
  x[i] += a[i] * b[i];
}

// OPM: pr f32 [12288][64] (pairs (i_feat, j_feat)) -> Ab/Bb bf16 [6144][64]
//   Ab[(pos*32+feat)][m] = pr[(m*192+pos)][2*feat]   (left operand rows)
//   Bb[(pos*32+feat)][m] = pr[(m*192+pos)][2*feat+1] (right operand, Bt layout)
__global__ void k_opm_split(const float* __restrict__ pr,
                            unsigned short* __restrict__ Ab,
                            unsigned short* __restrict__ Bb) {
  long long i = (long long)blockIdx.x * blockDim.x + threadIdx.x;
  if (i >= 192LL * 32 * 64) return;
  int m = (int)(i & 63);
  long long j = i >> 6;
  int feat = (int)(j & 31);
  int pos  = (int)(j >> 5);
  const float* s = pr + ((long long)m * 192 + pos) * 64;
  long long o = ((long long)pos * 32 + feat) * 64 + m;
  Ab[o] = f2bf(s[feat * 2]);
  Bb[o] = f2bf(s[feat * 2 + 1]);
}

// ---------------------------------------------------------------------------
// Host orchestration
// ---------------------------------------------------------------------------
enum {
  IN_X1D = 0, IN_X2D = 1,
  R_LNG = 2, R_LNB = 3, R_QKVW = 4, R_X2DW = 5, R_GATEW = 6, R_GATEB = 7, R_FINW = 8, R_FINB = 9,
  C_LNG = 10, C_LNB = 11, C_QKVW = 12, C_GATEW = 13, C_GATEB = 14, C_FINW = 15, C_FINB = 16,
  RT_LNG = 17, RT_LNB = 18, RT_L1W = 19, RT_L1B = 20, RT_L2W = 21, RT_L2B = 22,
  O_LNG = 23, O_LNB = 24, O_PROJW = 25, O_PROJB = 26, O_FINW = 27, O_FINB = 28,
  TO_BASE = 29, TI_BASE = 45,   // tri_p field order, see round-0 notes
  TS_BASE = 61, TE_BASE = 69,   // ta_p:  ln_g,ln_b,qkv_w,bias_w,gate_w,gate_b,out_w,out_b
  PT_LNG = 77, PT_LNB = 78, PT_L1W = 79, PT_L1B = 80, PT_L2W = 81, PT_L2B = 82
};

extern "C" void kernel_launch(void* const* d_in, const int* in_sizes, int n_in,
                              void* d_out, int out_size, void* d_ws, size_t ws_size,
                              hipStream_t stream) {
  (void)in_sizes; (void)n_in; (void)out_size; (void)ws_size;
  const float SCALE = 0.17677669529663687f;  // 1/sqrt(32)

  float* x1 = (float*)d_out;                 // running x1d [12288][256]
  float* x2 = x1 + 3145728;                  // running x2d [36864][128]
  hipMemcpyAsync(x1, d_in[IN_X1D], 3145728ull * 4, hipMemcpyDeviceToDevice, stream);
  hipMemcpyAsync(x2, d_in[IN_X2D], 4718592ull * 4, hipMemcpyDeviceToDevice, stream);

  // bump allocator over d_ws
  char* wsb = (char*)d_ws;
  size_t ofs = 0;
  auto alloc = [&](size_t bytes) -> char* {
    char* p = wsb + ofs; ofs += (bytes + 255) & ~size_t(255); return p;
  };
  auto aus  = [&](size_t n) -> unsigned short* { return (unsigned short*)alloc(n * 2); };
  auto af32 = [&](size_t n) -> float*          { return (float*)alloc(n * 4); };

  auto F = [&](int idx) -> const float* { return (const float*)d_in[idx]; };
  auto cvt = [&](int idx, int K, int N) -> const unsigned short* {
    unsigned short* t = aus((size_t)K * N);
    long long n = (long long)K * N;
    k_conv_t<<<dim3((unsigned)((n + 255) / 256)), 256, 0, stream>>>(F(idx), t, K, N);
    return t;
  };

  // ---- weights: transpose-convert to bf16 [N][K] -------------------------
  const unsigned short* r_qkv  = cvt(R_QKVW, 256, 768);
  const unsigned short* r_gate = cvt(R_GATEW, 256, 256);
  const unsigned short* r_fin  = cvt(R_FINW, 256, 256);
  const unsigned short* c_qkv  = cvt(C_QKVW, 256, 768);
  const unsigned short* c_gate = cvt(C_GATEW, 256, 256);
  const unsigned short* c_fin  = cvt(C_FINW, 256, 256);
  const unsigned short* rt_l1  = cvt(RT_L1W, 256, 1024);
  const unsigned short* rt_l2  = cvt(RT_L2W, 1024, 256);
  const unsigned short* o_proj = cvt(O_PROJW, 256, 64);
  const unsigned short* o_fin  = cvt(O_FINW, 1024, 128);
  int tmb[2] = {TO_BASE, TI_BASE};
  const unsigned short* tm_w[2][6];
  for (int d = 0; d < 2; ++d) {
    tm_w[d][0] = cvt(tmb[d] + 4, 128, 128);   // l1i
    tm_w[d][1] = cvt(tmb[d] + 6, 128, 128);   // l1j
    tm_w[d][2] = cvt(tmb[d] + 8, 128, 128);   // l1is
    tm_w[d][3] = cvt(tmb[d] + 10, 128, 128);  // l1js
    tm_w[d][4] = cvt(tmb[d] + 12, 128, 128);  // proj
    tm_w[d][5] = cvt(tmb[d] + 14, 128, 128);  // sigm
  }
  int tab[2] = {TS_BASE, TE_BASE};
  const unsigned short* ta_qkv[2]  = {cvt(TS_BASE + 2, 128, 768), cvt(TE_BASE + 2, 128, 768)};
  const unsigned short* ta_gate[2] = {cvt(TS_BASE + 4, 128, 256), cvt(TE_BASE + 4, 128, 256)};
  const unsigned short* ta_out[2]  = {cvt(TS_BASE + 6, 256, 128), cvt(TE_BASE + 6, 256, 128)};
  const unsigned short* pt_l1 = cvt(PT_L1W, 128, 512);
  const unsigned short* pt_l2 = cvt(PT_L2W, 512, 128);

  // ---- scratch (sized for tri-attention worst case, reused per phase) ----
  unsigned short* bLN  = aus(4718592);    // LN output bf16
  unsigned short* bQKV = aus(28311552);   // qkv / transition hidden (bf16)
  unsigned short* bQ   = aus(9437184);    // Q / I_t / opm-A
  unsigned short* bK   = aus(9437184);    // K / J_t / opm-B
  unsigned short* bVT  = aus(9437184);    // V transposed
  float*          bGATE= af32(9437184);   // gate f32 / tri A1,S1 / opm pr
  unsigned short* bAFF = aus(56623104);   // attention logits/weights (bf16) / x2buf / E
  float*          bAO  = af32(9437184);   // attention output f32 / tri A2,S2
  unsigned short* bMRG = aus(9437184);    // merged gated heads bf16 / Eb
  float*          bBIAS= af32(294912);    // pair bias [R2][8]

  auto gemm = [&](const unsigned short* A, const unsigned short* B, float* Cf,
                  unsigned short* Cb, const float* bias, const float* resid,
                  int M, int N, int K, int nb, long long sA, long long sB, long long sC,
                  int act, float scale, int wmode) {
    if ((M & 63) == 0 && (N & 63) == 0) {
      long long blocks = (long long)(M / 64) * (N / 64) * nb;
      k_gemm_wmma_tiled<<<dim3((unsigned)blocks), 256, 0, stream>>>(
          A, B, Cf, Cb, bias, resid, M, N, K, nb, sA, sB, sC, act, scale, wmode);
    } else {
      long long tiles = (long long)(M / 16) * (N / 16) * nb;
      k_gemm_wmma<<<dim3((unsigned)((tiles + 7) / 8)), 256, 0, stream>>>(
          A, B, Cf, Cb, bias, resid, M, N, K, nb, sA, sB, sC, act, scale, wmode);
    }
  };
  auto ln = [&](const float* x, unsigned short* o, int gi, int bi, int rows, int C) {
    k_ln_rows<<<dim3((unsigned)((rows + 7) / 8)), 256, 0, stream>>>(x, o, F(gi), F(bi), rows, C);
  };
  // split -> QK^T -> softmax -> PV -> gate-merge (shared by all attentions)
  auto attn_core = [&](int P, int rows, int mode, float qscale,
                       const float* smBias, int rs, int cs) {
    int nb = P * 8;
    long long th = (long long)P * rows * 256;
    k_split_heads<<<dim3((unsigned)((th + 255) / 256)), 256, 0, stream>>>(
        bQKV, bQ, bK, bVT, P, rows, mode, qscale);
    gemm(bQ, bK, nullptr, bAFF, nullptr, nullptr, rows, rows, 32, nb,
         (long long)rows * 32, (long long)rows * 32, (long long)rows * rows, 0, 1.f, 0);
    long long waves = (long long)nb * rows;
    k_softmax<<<dim3((unsigned)((waves + 7) / 8)), 256, 0, stream>>>(
        bAFF, smBias, nb, rows, rows, rs, cs);
    gemm(bAFF, bVT, bAO, nullptr, nullptr, nullptr, rows, 32, rows, nb,
         (long long)rows * rows, 32LL * rows, (long long)rows * 32, 0, 1.f, 0);
    long long tm = (long long)nb * rows * 32;
    k_merge_gate<<<dim3((unsigned)((tm + 255) / 256)), 256, 0, stream>>>(
        bAO, bGATE, bMRG, P, rows, mode);
  };

  // ===== 1. row attention (with pair bias) ================================
  ln(x1, bLN, R_LNG, R_LNB, 12288, 256);
  gemm(bLN, r_qkv, nullptr, bQKV, nullptr, nullptr, 12288, 768, 256, 1, 0, 0, 0, 0, 1.f, 0);
  gemm(bLN, r_gate, bGATE, nullptr, F(R_GATEB), nullptr, 12288, 256, 256, 1, 0, 0, 0, 2, 1.f, 0);
  k_dot8_f32<<<dim3((unsigned)((36864LL * 8 + 255) / 256)), 256, 0, stream>>>(
      x2, F(R_X2DW), bBIAS, 36864, 128, SCALE);
  attn_core(64, 192, 0, 1.0f, bBIAS, 1536, 8);
  gemm(bMRG, r_fin, x1, nullptr, F(R_FINB), x1, 12288, 256, 256, 1, 0, 0, 0, 0, 1.f, 0);

  // ===== 2. column attention =============================================
  ln(x1, bLN, C_LNG, C_LNB, 12288, 256);
  gemm(bLN, c_qkv, nullptr, bQKV, nullptr, nullptr, 12288, 768, 256, 1, 0, 0, 0, 0, 1.f, 0);
  gemm(bLN, c_gate, bGATE, nullptr, F(C_GATEB), nullptr, 12288, 256, 256, 1, 0, 0, 0, 2, 1.f, 0);
  attn_core(192, 64, 1, SCALE, nullptr, 0, 0);
  gemm(bMRG, c_fin, x1, nullptr, F(C_FINB), x1, 12288, 256, 256, 1, 0, 0, 0, 0, 1.f, 0);

  // ===== 3. recycle (1d) transition ======================================
  ln(x1, bLN, RT_LNG, RT_LNB, 12288, 256);
  gemm(bLN, rt_l1, nullptr, bQKV, F(RT_L1B), nullptr, 12288, 1024, 256, 1, 0, 0, 0, 1, 1.f, 0);
  gemm(bQKV, rt_l2, x1, nullptr, F(RT_L2B), x1, 12288, 256, 1024, 1, 0, 0, 0, 0, 1.f, 0);

  // ===== 4. outer product mean ===========================================
  ln(x1, bLN, O_LNG, O_LNB, 12288, 256);
  gemm(bLN, o_proj, bGATE, nullptr, F(O_PROJB), nullptr, 12288, 64, 256, 1, 0, 0, 0, 0, 1.f, 0);
  k_opm_split<<<dim3((unsigned)((192LL * 32 * 64 + 255) / 256)), 256, 0, stream>>>(bGATE, bQ, bK);
  gemm(bQ, bK, nullptr, bAFF, nullptr, nullptr, 6144, 6144, 64, 1, 0, 0, 0, 0, 1.f / 64.f, 1);
  gemm(bAFF, o_fin, x2, nullptr, F(O_FINB), x2, 36864, 128, 1024, 1, 0, 0, 0, 0, 1.f, 0);

  // ===== 5/6. triangle multiplication (outgoing d=0, incoming d=1) ========
  auto tri_mul = [&](int d, int mode) {
    int base = tmb[d];
    ln(x2, bLN, base + 0, base + 1, 36864, 128);
    float* A1 = bGATE;           float* S1 = bGATE + 4718592;
    float* A2 = bAO;             float* S2 = bAO + 4718592;
    gemm(bLN, tm_w[d][0], A1, nullptr, F(base + 5), nullptr, 36864, 128, 128, 1, 0, 0, 0, 0, 1.f, 0);
    gemm(bLN, tm_w[d][2], S1, nullptr, F(base + 9), nullptr, 36864, 128, 128, 1, 0, 0, 0, 2, 1.f, 0);
    gemm(bLN, tm_w[d][1], A2, nullptr, F(base + 7), nullptr, 36864, 128, 128, 1, 0, 0, 0, 0, 1.f, 0);
    gemm(bLN, tm_w[d][3], S2, nullptr, F(base + 11), nullptr, 36864, 128, 128, 1, 0, 0, 0, 2, 1.f, 0);
    long long tot = 36864LL * 128;
    k_mul_t<<<dim3((unsigned)((tot + 255) / 256)), 256, 0, stream>>>(A1, S1, bQ, mode);
    k_mul_t<<<dim3((unsigned)((tot + 255) / 256)), 256, 0, stream>>>(A2, S2, bK, mode);
    float* E = (float*)bAFF;     // einsum result f32 [36864][128] via wmode 2
    gemm(bQ, bK, E, nullptr, nullptr, nullptr, 192, 192, 192, 128, 36864, 36864, 0, 0, 1.f, 2);
    ln(E, bMRG, base + 2, base + 3, 36864, 128);
    gemm(bMRG, tm_w[d][4], A1, nullptr, F(base + 13), nullptr, 36864, 128, 128, 1, 0, 0, 0, 0, 1.f, 0);
    gemm(bLN,  tm_w[d][5], A2, nullptr, F(base + 15), nullptr, 36864, 128, 128, 1, 0, 0, 0, 2, 1.f, 0);
    k_ew_mac<<<dim3((unsigned)((tot + 255) / 256)), 256, 0, stream>>>(x2, A1, A2, tot);
  };
  tri_mul(0, 0);
  tri_mul(1, 1);

  // ===== 7/8. triangle attention (starting d=0, ending d=1) ===============
  auto tri_attn = [&](int d, int mode, int rs, int cs) {
    int base = tab[d];
    ln(x2, bLN, base + 0, base + 1, 36864, 128);
    gemm(bLN, ta_qkv[d], nullptr, bQKV, nullptr, nullptr, 36864, 768, 128, 1, 0, 0, 0, 0, 1.f, 0);
    gemm(bLN, ta_gate[d], bGATE, nullptr, F(base + 5), nullptr, 36864, 256, 128, 1, 0, 0, 0, 2, 1.f, 0);
    k_dot8_bf16<<<dim3((unsigned)((36864LL * 8 + 255) / 256)), 256, 0, stream>>>(
        bLN, F(base + 3), bBIAS, 36864, 128, 1.f);
    attn_core(192, 192, mode, SCALE, bBIAS, rs, cs);
    gemm(bMRG, ta_out[d], x2, nullptr, F(base + 7), x2, 36864, 128, 256, 1, 0, 0, 0, 0, 1.f, 0);
  };
  tri_attn(0, 0, 8, 1536);    // starting: bias[k,j,h] -> row(j)*8 + col(k)*1536
  tri_attn(1, 1, 1536, 8);    // ending:   bias[i,k,h] -> row(i)*1536 + col(k)*8

  // ===== 9. pair transition ==============================================
  ln(x2, bLN, PT_LNG, PT_LNB, 36864, 128);
  gemm(bLN, pt_l1, nullptr, bQKV, F(PT_L1B), nullptr, 36864, 512, 128, 1, 0, 0, 0, 1, 1.f, 0);
  gemm(bQKV, pt_l2, x2, nullptr, F(PT_L2B), x2, 36864, 128, 512, 1, 0, 0, 0, 0, 1.f, 0);
}
